// MultiHeadedAttention_43422119363254
// MI455X (gfx1250) — matrix operand order repaired and checked by reading, not compiled, not run on hardware
//
#include <hip/hip_runtime.h>
#include <hip/hip_bf16.h>
#include <stdint.h>

// MI455X / gfx1250, wave32. f16 WMMA (fp32 accumulate) everywhere:
// memory-bound kernel (512MB p_attn output dominates), so we take the
// chip's fast matrix path and keep all softmax math in fp32.
// This round: weight-slab staging in qkv_gemm goes through the Tensor Data
// Mover (tensor_load_to_lds + s_wait_tensorcnt), double-buffered in LDS.

typedef __attribute__((ext_vector_type(16))) _Float16 v16h;
typedef __attribute__((ext_vector_type(8)))  _Float16 v8h;
typedef __attribute__((ext_vector_type(8)))  float    v8f;
typedef __attribute__((ext_vector_type(4)))  float    v4f;
typedef __attribute__((ext_vector_type(4)))  unsigned int v4u;
typedef __attribute__((ext_vector_type(8)))  int      v8i;
typedef __attribute__((ext_vector_type(4)))  int      v4i;

#if __has_builtin(__builtin_amdgcn_tensor_load_to_lds)
#define HAVE_TDM 1
#else
#define HAVE_TDM 0
#endif

__device__ __forceinline__ v8f wmma16(v16h a, v16h b, v8f c) {
    // D = A(16x32 f16) * B(32x16 f16) + C(16x16 f32)
    return __builtin_amdgcn_wmma_f32_16x16x32_f16(
        /*neg_a=*/false, a, /*neg_b=*/false, b,
        /*c_mod=*/(short)0, c, /*reuse_a=*/false, /*reuse_b=*/false);
}

// A-fragment (16x32, f16, row-major source A[m][k], ld in halves).
// Lane l: row m = l&15, half-sel h = l>>4. Needs K = {8h..8h+7, 16+8h..16+8h+7}.
__device__ __forceinline__ v16h ldfragA(const _Float16* base, int ld) {
    int l = threadIdx.x & 31;
    const _Float16* p = base + (size_t)(l & 15) * ld + 8 * (l >> 4);
    v8h lo = *(const v8h*)(p);
    v8h hi = *(const v8h*)(p + 16);
    return __builtin_shufflevector(lo, hi, 0,1,2,3,4,5,6,7,8,9,10,11,12,13,14,15);
}

// B-fragment (32x16, f16) from an [n][k] (column-of-B per row) layout.
// Lane l: col n = l&15, needs K = {8h..8h+7, 16+8h..16+8h+7}, h = l>>4.
__device__ __forceinline__ v16h ldfragB(const _Float16* base, int ld) {
    int l = threadIdx.x & 31;
    const _Float16* p = base + (size_t)(l & 15) * ld + 8 * (l >> 4);
    v8h lo = *(const v8h*)(p);
    v8h hi = *(const v8h*)(p + 16);
    return __builtin_shufflevector(lo, hi, 0,1,2,3,4,5,6,7,8,9,10,11,12,13,14,15);
}

// A-fragment with on-the-fly fp32 -> f16 conversion (for raw inputs).
__device__ __forceinline__ v16h ldfragA_f32(const float* base, int ld) {
    int l = threadIdx.x & 31;
    const float* p = base + (size_t)(l & 15) * ld + 8 * (l >> 4);
    v4f f0 = *(const v4f*)(p);
    v4f f1 = *(const v4f*)(p + 4);
    v4f f2 = *(const v4f*)(p + 16);
    v4f f3 = *(const v4f*)(p + 20);
    v16h out;
#pragma unroll
    for (int i = 0; i < 4; ++i) {
        out[i]      = (_Float16)f0[i];
        out[4 + i]  = (_Float16)f1[i];
        out[8 + i]  = (_Float16)f2[i];
        out[12 + i] = (_Float16)f3[i];
    }
    return out;
}

// Low 32 bits of a flat shared-memory address = LDS byte offset (AS3 value).
__device__ __forceinline__ unsigned lds_off_of(const void* p) {
    return (unsigned)(size_t)p;
}

#if HAVE_TDM
// TDM: DMA a 64-row x 32-half tile of a row-major [1024][1024] f16 matrix
// into LDS. data_size=2B. LDS padding: after every 16 DWORDs (one 64B row)
// insert 4 DWORDs -> LDS row stride = 80B = 40 halves (bank-conflict-free
// ds_load_b128 B-fragments: lane r starts at dword 20*r mod 64).
__device__ __forceinline__ void tdm_load_wt_tile(unsigned lds_addr,
                                                 const _Float16* gsrc) {
    unsigned long long ga = (unsigned long long)(size_t)gsrc;
    v4u g0;
    g0[0] = 1u;                                   // count=1 (valid), user mode
    g0[1] = lds_addr;                             // lds_addr [63:32]
    g0[2] = (unsigned)(ga & 0xFFFFFFFFu);         // global_addr lo
    g0[3] = (unsigned)((ga >> 32) & 0x01FFFFFFu)  // global_addr hi (57-bit)
          | (2u << 30);                           // type=2 ("image")
    v8i g1;
    g1[0] = (1 << 16)        // data_size = 1 -> 2-byte elements
          | (1 << 20)        // pad_enable
          | (3 << 22)        // pad_interval = 16 DWORDs (one tile row)
          | (3 << 25);       // pad_amount   = 4 DWORDs  (row stride 80B)
    g1[1] = 1024 << 16;      // tensor_dim0 = 1024 (bits[79:48], lo16 here)
    g1[2] = 1024 << 16;      // dim0 hi16 = 0 | tensor_dim1 = 1024 lo16
    g1[3] = 32 << 16;        // dim1 hi16 = 0 | tile_dim0 = 32
    g1[4] = 64;              // tile_dim1 = 64 | tile_dim2 = 0
    g1[5] = 1024;            // tensor_dim0_stride = 1024 elements (lo32)
    g1[6] = 0;               // dim0_stride hi16 | tensor_dim1_stride lo16
    g1[7] = 0;
    v4i z4 = {};
#if __clang_major__ >= 23
    v8i z8 = {};
    __builtin_amdgcn_tensor_load_to_lds(g0, g1, z4, z4, z8, 0);
#else
    __builtin_amdgcn_tensor_load_to_lds(g0, g1, z4, z4, 0);
#endif
}
#endif  // HAVE_TDM

// ---------------------------------------------------------------------------
// Kernel 1: W (1024x1024 fp32, [k][n]) -> Wt (f16, [n][k]) via LDS transpose.
// ---------------------------------------------------------------------------
__global__ __launch_bounds__(256) void transposeW(const float* __restrict__ W,
                                                  _Float16* __restrict__ Wt) {
    __shared__ float tile[32][33];
    int x = threadIdx.x & 31, y = threadIdx.x >> 5;
    int n0 = blockIdx.x * 32, k0 = blockIdx.y * 32;
#pragma unroll
    for (int yy = y; yy < 32; yy += 8)
        tile[yy][x] = W[(size_t)(k0 + yy) * 1024 + n0 + x];
    __syncthreads();
#pragma unroll
    for (int yy = y; yy < 32; yy += 8)
        Wt[(size_t)(n0 + yy) * 1024 + k0 + x] = (_Float16)tile[x][yy];
}

// ---------------------------------------------------------------------------
// Kernel 2: fused QKV projection. grid = (M/128, N/64, 3). 8 waves/WG,
// each wave computes a 32x32 tile (2x2 WMMA accumulators), K-loop step 32.
// The 64x32 Wt slab for each K-step is DMA'd into LDS by the Tensor Data
// Mover (wave 0 issues the descriptor; double-buffered so the DMA for step
// k+1 overlaps the WMMAs on step k; s_wait_tensorcnt + barrier publishes it).
// Outputs: q_h, k_h row-major [B*T][1024] f16; V stored per-head transposed
// vT[(b*16+h)*64+d][t] f16 so the PV GEMM B-fragments are contiguous.
// ---------------------------------------------------------------------------
#define WSTR 40  // padded LDS row stride in halves (80B), set by TDM pad fields

__global__ __launch_bounds__(256) void qkv_gemm(
    const float* __restrict__ Xq, const float* __restrict__ Xk,
    const float* __restrict__ Xv, const _Float16* __restrict__ WtAll,
    const float* __restrict__ bq, const float* __restrict__ bk,
    const float* __restrict__ bv,
    _Float16* __restrict__ qh, _Float16* __restrict__ kh,
    _Float16* __restrict__ vT) {
    const int z = blockIdx.z;
    const float*    X    = (z == 0) ? Xq : (z == 1) ? Xk : Xv;
    const _Float16* Wt   = WtAll + (size_t)z * 1024 * 1024;
    const float*    bias = (z == 0) ? bq : (z == 1) ? bk : bv;

    const int M0 = blockIdx.x * 128, N0 = blockIdx.y * 64;
    const int w = threadIdx.x >> 5, lane = threadIdx.x & 31;
    const int mbase = M0 + (w & 3) * 32;
    const int nbase = N0 + (w >> 2) * 32;
    const int nloc  = (w >> 2) * 32;  // this wave's B rows inside the LDS slab

    __shared__ __attribute__((aligned(16))) _Float16 wbuf[2][64 * WSTR + 16];

    v8f acc[2][2] = {};
    int bf = 0;
#if HAVE_TDM
    if (threadIdx.x < 32)
        tdm_load_wt_tile(lds_off_of(&wbuf[0][0]), Wt + (size_t)N0 * 1024);
#endif
    for (int k0 = 0; k0 < 1024; k0 += 32) {
#if HAVE_TDM
        if (threadIdx.x < 32)
            __builtin_amdgcn_s_wait_tensorcnt(0);  // slab bf landed in LDS
        __syncthreads();                            // publish to all waves
        if (k0 + 32 < 1024 && threadIdx.x < 32)     // overlap next DMA w/ math
            tdm_load_wt_tile(lds_off_of(&wbuf[bf ^ 1][0]),
                             Wt + (size_t)N0 * 1024 + k0 + 32);
#else
        // Fallback: cooperative synchronous staging (same LDS layout).
        __syncthreads();
        {
            int row = threadIdx.x >> 2, ch = (threadIdx.x & 3) * 8;
            *(v8h*)&wbuf[bf][row * WSTR + ch] =
                *(const v8h*)(Wt + (size_t)(N0 + row) * 1024 + k0 + ch);
        }
        __syncthreads();
#endif
        if (k0 + 32 < 1024)
            __builtin_prefetch(X + (size_t)mbase * 1024 + k0 + 32, 0, 1);
        v16h a0 = ldfragA_f32(X + (size_t)mbase * 1024 + k0, 1024);
        v16h a1 = ldfragA_f32(X + (size_t)(mbase + 16) * 1024 + k0, 1024);
        v16h b0 = ldfragB(&wbuf[bf][nloc * WSTR], WSTR);
        v16h b1 = ldfragB(&wbuf[bf][(nloc + 16) * WSTR], WSTR);
        acc[0][0] = wmma16(a0, b0, acc[0][0]);
        acc[0][1] = wmma16(a0, b1, acc[0][1]);
        acc[1][0] = wmma16(a1, b0, acc[1][0]);
        acc[1][1] = wmma16(a1, b1, acc[1][1]);
        bf ^= 1;
    }

    const int n = lane & 15, hp = lane >> 4;
#pragma unroll
    for (int mt = 0; mt < 2; ++mt) {
#pragma unroll
        for (int nt = 0; nt < 2; ++nt) {
            const int ncol = nbase + nt * 16 + n;
            const float bb = bias[ncol];
            const int mrow = mbase + mt * 16 + 8 * hp;  // 8 consecutive rows/lane
            if (z < 2) {
                _Float16* o = (z == 0 ? qh : kh) + (size_t)mrow * 1024 + ncol;
#pragma unroll
                for (int j = 0; j < 8; ++j)
                    o[(size_t)j * 1024] = (_Float16)(acc[mt][nt][j] + bb);
            } else {
                const int head = ncol >> 6, d = ncol & 63;
                const int batch = mrow >> 10, t = mrow & 1023;
                v8h pk;
#pragma unroll
                for (int j = 0; j < 8; ++j)
                    pk[j] = (_Float16)(acc[mt][nt][j] + bb);
                *(v8h*)(vT + (size_t)((batch * 16 + head) * 64 + d) * 1024 + t) = pk;
            }
        }
    }
}

// ---------------------------------------------------------------------------
// Kernel 3: attention. grid = (T/16, H, B), 8 waves/WG.
// Phase 1: each wave computes a 16x128 score strip (QK^T) -> LDS (f16).
// Phase 2: fp32 softmax per row (wave32 shfl reductions); p_attn streamed
//          out fp32 with non-temporal stores; normalized P kept in LDS (f16).
// Phase 3: waves 0..3 compute x = P @ V_head (16x16 tile each, full K=1024).
// ---------------------------------------------------------------------------
#define PSTR 1032  // 1024 + 8 halves padding (keeps 16B alignment, spreads banks)

__global__ __launch_bounds__(256) void attn(
    const _Float16* __restrict__ qh, const _Float16* __restrict__ kh,
    const _Float16* __restrict__ vT, const unsigned char* __restrict__ mask,
    float* __restrict__ o_attn, float* __restrict__ p_attn) {
    __shared__ __attribute__((aligned(16))) _Float16 psc[16 * PSTR];

    const int b = blockIdx.z, h = blockIdx.y, t0 = blockIdx.x * 16;
    const int w = threadIdx.x >> 5, lane = threadIdx.x & 31;
    const int n = lane & 15, hp = lane >> 4;
    const float scale = 0.125f;  // 1/sqrt(64)

    // Q tile (16 x 64) A-fragments for the two K-steps of the head dim.
    const _Float16* qbase = qh + (size_t)(b * 1024 + t0) * 1024 + h * 64;
    const v16h aq0 = ldfragA(qbase, 1024);
    const v16h aq1 = ldfragA(qbase + 32, 1024);

    // Scores: wave w owns columns [w*128, w*128+128).
    const _Float16* kbase = kh + (size_t)(b * 1024) * 1024 + h * 64;
    for (int ct = 0; ct < 8; ++ct) {
        const int s0 = w * 128 + ct * 16;
        v8f acc = {};
        acc = wmma16(aq0, ldfragB(kbase + (size_t)s0 * 1024, 1024), acc);
        acc = wmma16(aq1, ldfragB(kbase + (size_t)s0 * 1024 + 32, 1024), acc);
        const unsigned char mk = mask[b * 1024 + s0 + n];
#pragma unroll
        for (int j = 0; j < 8; ++j) {
            const float v = mk ? acc[j] * scale : -60000.0f;  // finite in f16; exp -> 0
            psc[(j + 8 * hp) * PSTR + s0 + n] = (_Float16)v;
        }
    }
    __syncthreads();

    // Softmax: wave w handles rows 2w and 2w+1 (fp32 math).
    for (int rr = 0; rr < 2; ++rr) {
        const int r = 2 * w + rr;
        float mx = -3.0e38f;
        for (int c = lane; c < 1024; c += 32)
            mx = fmaxf(mx, (float)psc[r * PSTR + c]);
#pragma unroll
        for (int off = 16; off; off >>= 1)
            mx = fmaxf(mx, __shfl_xor(mx, off, 32));
        float sum = 0.0f;
        for (int c = lane; c < 1024; c += 32) {
            const float e = expf((float)psc[r * PSTR + c] - mx);
            sum += e;
            psc[r * PSTR + c] = (_Float16)e;
        }
#pragma unroll
        for (int off = 16; off; off >>= 1)
            sum += __shfl_xor(sum, off, 32);
        const float inv = 1.0f / sum;
        float* prow = p_attn + ((size_t)(b * 16 + h) * 1024 + (t0 + r)) * 1024;
        for (int c = lane; c < 1024; c += 32) {
            const float pv = (float)psc[r * PSTR + c] * inv;
            __builtin_nontemporal_store(pv, prow + c);  // 512MB stream: don't pollute L2
            psc[r * PSTR + c] = (_Float16)pv;
        }
    }
    __syncthreads();

    // PV: wave w<4 computes output columns d0 = w*16 over full K = 1024.
    if (w < 4) {
        const int d0 = w * 16;
        const _Float16* vbase = vT + ((size_t)(b * 16 + h) * 64 + d0) * 1024;
        v8f acc = {};
        for (int ks = 0; ks < 32; ++ks) {
            v16h ap = ldfragA(psc + ks * 32, PSTR);       // P from LDS
            v16h bvf = ldfragB(vbase + ks * 32, 1024);    // V (pre-transposed)
            acc = wmma16(ap, bvf, acc);
        }
        float* obase = o_attn + (size_t)(b * 1024 + t0) * 1024 + h * 64 + d0;
#pragma unroll
        for (int j = 0; j < 8; ++j)
            obase[(size_t)(j + 8 * hp) * 1024 + n] = acc[j];
    }
}

// ---------------------------------------------------------------------------
extern "C" void kernel_launch(void* const* d_in, const int* in_sizes, int n_in,
                              void* d_out, int out_size, void* d_ws, size_t ws_size,
                              hipStream_t stream) {
    const float* queries = (const float*)d_in[0];
    const float* keys    = (const float*)d_in[1];
    const float* values  = (const float*)d_in[2];
    const unsigned char* mask = (const unsigned char*)d_in[3];  // numpy bool = 1 byte
    const float* Wq = (const float*)d_in[4];
    const float* bq = (const float*)d_in[5];
    const float* Wk = (const float*)d_in[6];
    const float* bk = (const float*)d_in[7];
    const float* Wv = (const float*)d_in[8];
    const float* bv = (const float*)d_in[9];

    // Workspace layout (f16 elements): 3x Wt (1M each) + q_h/k_h/vT (8M each)
    // = 27M halves = 54 MB.
    _Float16* ws = (_Float16*)d_ws;
    const size_t WSZ = (size_t)1024 * 1024;
    const size_t QSZ = (size_t)8192 * 1024;
    _Float16* wt  = ws;
    _Float16* qhb = ws + 3 * WSZ;
    _Float16* khb = qhb + QSZ;
    _Float16* vTb = khb + QSZ;

    float* o_attn = (float*)d_out;                       // (8,1024,1024) fp32
    float* p_attn = o_attn + (size_t)8 * 1024 * 1024;    // (8,16,1024,1024) fp32

    transposeW<<<dim3(32, 32), 256, 0, stream>>>(Wq, wt);
    transposeW<<<dim3(32, 32), 256, 0, stream>>>(Wk, wt + WSZ);
    transposeW<<<dim3(32, 32), 256, 0, stream>>>(Wv, wt + 2 * WSZ);
    qkv_gemm<<<dim3(64, 16, 3), 256, 0, stream>>>(queries, keys, values, wt,
                                                  bq, bk, bv, qhb, khb, vTb);
    attn<<<dim3(64, 16, 8), 256, 0, stream>>>(qhb, khb, vTb, mask, o_attn, p_attn);

    (void)in_sizes; (void)n_in; (void)out_size; (void)ws_size;
}
